// MultiHeadAttention_76940044140953
// MI455X (gfx1250) — matrix operand order repaired and checked by reading, not compiled, not run on hardware
//
#include <hip/hip_runtime.h>

#define DIMSZ 1024
#define INNERSZ 1024
#define SEQ 2048
#define BATCH 4
#define NHEADS 16
#define HDIM 64

typedef __bf16 bf16;
typedef float v8f  __attribute__((ext_vector_type(8)));
typedef bf16  v16bf __attribute__((ext_vector_type(16)));
typedef bf16  v8bf  __attribute__((ext_vector_type(8)));

union V16U { v16bf v; v8bf h[2]; };

__device__ __forceinline__ v8f wmma_bf16f32(v16bf a, v16bf b, v8f c) {
  // D = A(16x32 bf16) * B(32x16 bf16) + C(16x16 f32)
  return __builtin_amdgcn_wmma_f32_16x16x32_bf16(
      false, a, false, b, (short)0, c, false, false);
}

// ---- CDNA5 async copy: global -> LDS, 16 B per lane (ASYNCcnt tracked) ----
__device__ __forceinline__ void async_copy_b128(const bf16* gsrc, bf16* ldst) {
  unsigned l = (unsigned)(unsigned long long)ldst;  // low 32 bits = LDS offset
  asm volatile("global_load_async_to_lds_b128 %0, %1, off"
               :: "v"(l), "v"(gsrc) : "memory");
}
__device__ __forceinline__ void wait_async0() {
  asm volatile("s_wait_asynccnt 0" ::: "memory");
}

// ---- 16-lane reductions: DPP16 fused into the ALU op (1 instr per step) --
__device__ __forceinline__ float rowmax16(float v) {
  asm("v_max_num_f32_dpp %0, %0, %0 row_ror:1 row_mask:0xf bank_mask:0xf bound_ctrl:1" : "+v"(v));
  asm("v_max_num_f32_dpp %0, %0, %0 row_ror:2 row_mask:0xf bank_mask:0xf bound_ctrl:1" : "+v"(v));
  asm("v_max_num_f32_dpp %0, %0, %0 row_ror:4 row_mask:0xf bank_mask:0xf bound_ctrl:1" : "+v"(v));
  asm("v_max_num_f32_dpp %0, %0, %0 row_ror:8 row_mask:0xf bank_mask:0xf bound_ctrl:1" : "+v"(v));
  return v;
}
__device__ __forceinline__ float rowsum16(float v) {
  asm("v_add_f32_dpp %0, %0, %0 row_ror:1 row_mask:0xf bank_mask:0xf bound_ctrl:1" : "+v"(v));
  asm("v_add_f32_dpp %0, %0, %0 row_ror:2 row_mask:0xf bank_mask:0xf bound_ctrl:1" : "+v"(v));
  asm("v_add_f32_dpp %0, %0, %0 row_ror:4 row_mask:0xf bank_mask:0xf bound_ctrl:1" : "+v"(v));
  asm("v_add_f32_dpp %0, %0, %0 row_ror:8 row_mask:0xf bank_mask:0xf bound_ctrl:1" : "+v"(v));
  return v;
}

// ---------------- LayerNorm (fp32 in) -> bf16 out -------------------------
__global__ __launch_bounds__(256) void ln_bf16_kernel(
    const float* __restrict__ x, const float* __restrict__ gamma,
    const float* __restrict__ beta, bf16* __restrict__ xn) {
  const int row = blockIdx.x;
  const float* xr = x + (size_t)row * DIMSZ;
  float s = 0.f, s2 = 0.f;
  for (int i = threadIdx.x; i < DIMSZ; i += 256) {
    float v = xr[i]; s += v; s2 += v * v;
  }
  __shared__ float rs[256], rs2[256];
  rs[threadIdx.x] = s; rs2[threadIdx.x] = s2;
  __syncthreads();
  for (int off = 128; off > 0; off >>= 1) {
    if (threadIdx.x < off) {
      rs[threadIdx.x]  += rs[threadIdx.x + off];
      rs2[threadIdx.x] += rs2[threadIdx.x + off];
    }
    __syncthreads();
  }
  const float mean = rs[0] * (1.0f / DIMSZ);
  const float var  = rs2[0] * (1.0f / DIMSZ) - mean * mean;
  const float inv  = rsqrtf(var + 1e-6f);
  bf16* xo = xn + (size_t)row * DIMSZ;
  for (int i = threadIdx.x; i < DIMSZ; i += 256)
    xo[i] = (bf16)((xr[i] - mean) * inv * gamma[i] + beta[i]);
}

// ---------------- transpose + cast: W[K][N] f32 -> W^T[N][K] bf16 ---------
__global__ __launch_bounds__(256) void cvtT_bf16_kernel(
    const float* __restrict__ in, bf16* __restrict__ out, int K, int N) {
  __shared__ bf16 t[32][33];
  const int k0 = blockIdx.y * 32, n0 = blockIdx.x * 32;
#pragma unroll
  for (int i = 0; i < 4; ++i) {
    int k = threadIdx.y + 8 * i;
    t[k][threadIdx.x] = (bf16)in[(size_t)(k0 + k) * N + n0 + threadIdx.x];
  }
  __syncthreads();
#pragma unroll
  for (int i = 0; i < 4; ++i) {
    int n = threadIdx.y + 8 * i;
    out[(size_t)(n0 + n) * K + k0 + threadIdx.x] = t[threadIdx.x][n];
  }
}

// ---------------- bf16 WMMA GEMM: C[M,N] = A[M,K] * (W^T[N,K])^T ----------
// 256 thr = 8 waves (4x2); block tile 128x128; wave tile 32x64; K-step 32;
// double-buffered LDS tiles filled by async-to-LDS, overlapped with WMMA.
enum { OUT_BF16 = 0, OUT_BF16_T = 1, OUT_F32_BIAS = 2 };

template <int MODE>
__global__ __launch_bounds__(256) void gemm_bf16_kernel(
    const bf16* __restrict__ A, const bf16* __restrict__ WT,
    bf16* __restrict__ outb, float* __restrict__ outf,
    const float* __restrict__ bias, int M, int N, int K) {
  __shared__ bf16 At[2][128][40];  // [m][k], pitch 40 halves
  __shared__ bf16 Bt[2][128][40];  // [n][k]
  const int tid = threadIdx.x;
  const int wave = tid >> 5, lane = tid & 31;
  const int lg = lane >> 4, ln = lane & 15;
  const int wm = wave >> 1, wn = wave & 1;
  const int m0 = blockIdx.y * 128, n0 = blockIdx.x * 128;
  // cooperative-load chunk coords (512 chunks of 8 halves per tile)
  const int r1 = tid >> 2,           c1 = (tid & 3) << 3;
  const int r2 = (tid + 256) >> 2,   c2 = ((tid + 256) & 3) << 3;

  auto issue = [&](int buf, int k0) {
    async_copy_b128(A  + (size_t)(m0 + r1) * K + k0 + c1, &At[buf][r1][c1]);
    async_copy_b128(A  + (size_t)(m0 + r2) * K + k0 + c2, &At[buf][r2][c2]);
    async_copy_b128(WT + (size_t)(n0 + r1) * K + k0 + c1, &Bt[buf][r1][c1]);
    async_copy_b128(WT + (size_t)(n0 + r2) * K + k0 + c2, &Bt[buf][r2][c2]);
  };

  v8f acc[2][4] = {};
  issue(0, 0);
  wait_async0();
  __syncthreads();
  const int nIter = K >> 5;
  for (int it = 0; it < nIter; ++it) {
    const int cur = it & 1;
    if (it + 1 < nIter) issue(cur ^ 1, (it + 1) << 5);  // prefetch next tile
    V16U af[2], bfr[4];
#pragma unroll
    for (int ms = 0; ms < 2; ++ms) {  // A frag: lane&15 = M row
      const bf16* rowp = &At[cur][wm * 32 + ms * 16 + ln][0];
      af[ms].h[0] = *(const v8bf*)(rowp + 16 * lg);
      af[ms].h[1] = *(const v8bf*)(rowp + 16 * lg + 8);
    }
#pragma unroll
    for (int ns = 0; ns < 4; ++ns) {  // B frag: lane&15 = N col
      const bf16* colp = &Bt[cur][wn * 64 + ns * 16 + ln][16 * lg];
      bfr[ns].h[0] = *(const v8bf*)(colp);
      bfr[ns].h[1] = *(const v8bf*)(colp + 8);
    }
#pragma unroll
    for (int ms = 0; ms < 2; ++ms)
#pragma unroll
      for (int ns = 0; ns < 4; ++ns)
        acc[ms][ns] = wmma_bf16f32(af[ms].v, bfr[ns].v, acc[ms][ns]);
    wait_async0();
    __syncthreads();
  }
#pragma unroll
  for (int ms = 0; ms < 2; ++ms)
#pragma unroll
    for (int ns = 0; ns < 4; ++ns) {
      const int row0 = m0 + wm * 32 + ms * 16 + 8 * lg;  // D row = row0 + r
      const int col  = n0 + wn * 64 + ns * 16 + ln;
      if (MODE == OUT_F32_BIAS) {
        float bv = bias[col];
#pragma unroll
        for (int r = 0; r < 8; ++r)
          outf[(size_t)(row0 + r) * N + col] = acc[ms][ns][r] + bv;
      } else if (MODE == OUT_BF16) {
#pragma unroll
        for (int r = 0; r < 8; ++r)
          outb[(size_t)(row0 + r) * N + col] = (bf16)acc[ms][ns][r];
      } else {  // OUT_BF16_T: [B,H,Dh,S]; row=b*SEQ+s, col=h*64+d
        union { uint4 u; bf16 e[8]; } pk;
#pragma unroll
        for (int r = 0; r < 8; ++r) pk.e[r] = (bf16)acc[ms][ns][r];
        size_t idx = (size_t)(row0 >> 11) * ((size_t)INNERSZ * SEQ)
                   + (size_t)col * SEQ + (row0 & (SEQ - 1));
        *(uint4*)(outb + idx) = pk.u;  // 8 consecutive s positions
      }
    }
}

// ---------------- Flash attention: 8 waves, 16 q-rows each ----------------
// Q,K: [B,S,H*Dh] bf16; VT: [B,H,Dh,S] bf16; Ctx: [B,S,H*Dh] bf16
__global__ __launch_bounds__(256) void attn_kernel(
    const bf16* __restrict__ Q, const bf16* __restrict__ Kd,
    const bf16* __restrict__ VT, bf16* __restrict__ Ctx) {
  __shared__ bf16 Kt[2][32][72];   // [token][dh]
  __shared__ bf16 VtT[2][64][40];  // [dh][token]
  __shared__ bf16 Pt[8][16][40];   // per-wave P tile [m][k]
  const int bb = blockIdx.z, h = blockIdx.y, qt = blockIdx.x;
  const int tid = threadIdx.x;
  const int wave = tid >> 5, lane = tid & 31;
  const int lg = lane >> 4, ln = lane & 15;
  const size_t hb = (size_t)bb * SEQ * INNERSZ + (size_t)h * HDIM;
  const size_t vb = ((size_t)bb * NHEADS + h) * HDIM * SEQ;
  const int q0 = qt * 128 + wave * 16;
  // cooperative-load chunk coords (256 chunks each)
  const int kr = tid >> 3, kc = (tid & 7) << 3;  // K tile 32x64
  const int vr = tid >> 2, vc = (tid & 3) << 3;  // V^T tile 64x32

  auto issue = [&](int buf, int key0) {
    async_copy_b128(Kd + hb + (size_t)(key0 + kr) * INNERSZ + kc, &Kt[buf][kr][kc]);
    async_copy_b128(VT + vb + (size_t)vr * SEQ + key0 + vc, &VtT[buf][vr][vc]);
  };

  V16U aq[2];
  const bf16* qrow = Q + hb + (size_t)(q0 + ln) * INNERSZ;
#pragma unroll
  for (int ks = 0; ks < 2; ++ks) {
    aq[ks].h[0] = *(const v8bf*)(qrow + ks * 32 + 16 * lg);
    aq[ks].h[1] = *(const v8bf*)(qrow + ks * 32 + 16 * lg + 8);
  }

  float mst[8], lst[8];
  v8f o[4] = {};
#pragma unroll
  for (int r = 0; r < 8; ++r) { mst[r] = -3.0e38f; lst[r] = 0.f; }
  const float scale = 0.125f;  // 1/sqrt(64)

  issue(0, 0);
  wait_async0();
  __syncthreads();
  const int nIter = SEQ >> 5;
  for (int kb = 0; kb < nIter; ++kb) {
    const int cur = kb & 1;
    if (kb + 1 < nIter) issue(cur ^ 1, (kb + 1) << 5);  // prefetch next tiles

    // S = Q * K^T : B[kd][n] = K[token n][dh kd] -> contiguous in Kt rows
    V16U bk[2][2];
#pragma unroll
    for (int ns = 0; ns < 2; ++ns)
#pragma unroll
      for (int ks = 0; ks < 2; ++ks) {
        const bf16* kp = &Kt[cur][ns * 16 + ln][ks * 32 + 16 * lg];
        bk[ns][ks].h[0] = *(const v8bf*)(kp);
        bk[ns][ks].h[1] = *(const v8bf*)(kp + 8);
      }
    v8f sc[2];
#pragma unroll
    for (int ns = 0; ns < 2; ++ns) {
      v8f c = {};
      c = wmma_bf16f32(aq[0].v, bk[ns][0].v, c);
      c = wmma_bf16f32(aq[1].v, bk[ns][1].v, c);
#pragma unroll
      for (int r = 0; r < 8; ++r) sc[ns][r] = c[r] * scale;
    }

    // online softmax (fused DPP16 rotate-reductions; wave32)
    float mnew[8], alpha[8];
#pragma unroll
    for (int r = 0; r < 8; ++r) {
      float v = rowmax16(fmaxf(sc[0][r], sc[1][r]));
      mnew[r]  = fmaxf(mst[r], v);
      alpha[r] = __expf(mst[r] - mnew[r]);
      mst[r]   = mnew[r];
    }
#pragma unroll
    for (int r = 0; r < 8; ++r) {
      float p0 = __expf(sc[0][r] - mnew[r]);
      float p1 = __expf(sc[1][r] - mnew[r]);
      Pt[wave][r + 8 * lg][ln]      = (bf16)p0;  // per-wave; DS is in-order
      Pt[wave][r + 8 * lg][16 + ln] = (bf16)p1;
      lst[r] = lst[r] * alpha[r] + rowsum16(p0 + p1);
    }
#pragma unroll
    for (int nd = 0; nd < 4; ++nd)
#pragma unroll
      for (int r = 0; r < 8; ++r) o[nd][r] *= alpha[r];

    // O += P(A-frag) * V ; B[kv][n] = V[token kv][dh n] = VtT[n][kv]
    V16U pa, bv[4];
    const bf16* pp = &Pt[wave][ln][0];
    pa.h[0] = *(const v8bf*)(pp + 16 * lg);
    pa.h[1] = *(const v8bf*)(pp + 16 * lg + 8);
#pragma unroll
    for (int nd = 0; nd < 4; ++nd) {
      const bf16* vp = &VtT[cur][nd * 16 + ln][16 * lg];
      bv[nd].h[0] = *(const v8bf*)(vp);
      bv[nd].h[1] = *(const v8bf*)(vp + 8);
    }
#pragma unroll
    for (int nd = 0; nd < 4; ++nd)
      o[nd] = wmma_bf16f32(pa.v, bv[nd].v, o[nd]);

    wait_async0();
    __syncthreads();
  }

  float invl[8];
#pragma unroll
  for (int r = 0; r < 8; ++r) invl[r] = 1.0f / lst[r];
#pragma unroll
  for (int nd = 0; nd < 4; ++nd)
#pragma unroll
    for (int r = 0; r < 8; ++r) {
      size_t idx = hb + (size_t)(q0 + r + 8 * lg) * INNERSZ + nd * 16 + ln;
      Ctx[idx] = (bf16)(o[nd][r] * invl[r]);
    }
}

// --------------------------------------------------------------------------
extern "C" void kernel_launch(void* const* d_in, const int* in_sizes, int n_in,
                              void* d_out, int out_size, void* d_ws, size_t ws_size,
                              hipStream_t stream) {
  const float* x     = (const float*)d_in[0];
  const float* gamma = (const float*)d_in[1];
  const float* beta  = (const float*)d_in[2];
  const float* Wq    = (const float*)d_in[3];
  const float* Wk    = (const float*)d_in[4];
  const float* Wv    = (const float*)d_in[5];
  const float* Wo    = (const float*)d_in[6];
  const float* bo    = (const float*)d_in[7];
  float* out = (float*)d_out;

  const size_t NTOK = (size_t)BATCH * SEQ;  // 8192
  // ws (bf16): xn | WqT,WkT,WvT,WoT | Q,K,V  = 36M bf16 = 72 MB; ctx reuses xn
  bf16* ws   = (bf16*)d_ws;
  bf16* xn   = ws;
  bf16* wqT  = xn  + NTOK * DIMSZ;
  bf16* wkT  = wqT + (size_t)DIMSZ * INNERSZ;
  bf16* wvT  = wkT + (size_t)DIMSZ * INNERSZ;
  bf16* woT  = wvT + (size_t)DIMSZ * INNERSZ;
  bf16* Qb   = woT + (size_t)DIMSZ * INNERSZ;
  bf16* Kb   = Qb  + NTOK * INNERSZ;
  bf16* VbT  = Kb  + NTOK * INNERSZ;  // [B,H,Dh,S]
  bf16* ctx  = xn;

  ln_bf16_kernel<<<(int)NTOK, 256, 0, stream>>>(x, gamma, beta, xn);

  dim3 tb(32, 8), tg(INNERSZ / 32, DIMSZ / 32);
  cvtT_bf16_kernel<<<tg, tb, 0, stream>>>(Wq, wqT, DIMSZ, INNERSZ);
  cvtT_bf16_kernel<<<tg, tb, 0, stream>>>(Wk, wkT, DIMSZ, INNERSZ);
  cvtT_bf16_kernel<<<tg, tb, 0, stream>>>(Wv, wvT, DIMSZ, INNERSZ);
  cvtT_bf16_kernel<<<tg, tb, 0, stream>>>(Wo, woT, INNERSZ, DIMSZ);

  dim3 gp(INNERSZ / 128, (unsigned)(NTOK / 128));
  gemm_bf16_kernel<OUT_BF16><<<gp, 256, 0, stream>>>(
      xn, wqT, Qb, nullptr, nullptr, (int)NTOK, INNERSZ, DIMSZ);
  gemm_bf16_kernel<OUT_BF16><<<gp, 256, 0, stream>>>(
      xn, wkT, Kb, nullptr, nullptr, (int)NTOK, INNERSZ, DIMSZ);
  gemm_bf16_kernel<OUT_BF16_T><<<gp, 256, 0, stream>>>(
      xn, wvT, VbT, nullptr, nullptr, (int)NTOK, INNERSZ, DIMSZ);

  attn_kernel<<<dim3(SEQ / 128, NHEADS, BATCH), 256, 0, stream>>>(Qb, Kb, VbT, ctx);

  gemm_bf16_kernel<OUT_F32_BIAS><<<dim3(DIMSZ / 128, (unsigned)(NTOK / 128)), 256, 0, stream>>>(
      ctx, woT, nullptr, out, bo, (int)NTOK, DIMSZ, INNERSZ);
}
